// Conv2DManual_40407052320931
// MI455X (gfx1250) — matrix operand order repaired and checked by reading, not compiled, not run on hardware
//
#include <hip/hip_runtime.h>

// ---------------------------------------------------------------------------
// Conv2D 3x3 (N=32, 128->256, 56x56, pad 1) as implicit GEMM on CDNA5 WMMA.
//
// Fast path (uses d_ws, 28.2 MB):
//   prep_weight: w[co][ci][r][s] f32 -> Wp[co][tap][ci] bf16   (0.59 MB)
//   prep_x     : x NCHW f32      -> Xp[n][ci][58][58]  bf16 zero-padded (27.6 MB)
//   conv_main  : K ordered k = tap*128 + ci  (tap/ci split = shifts only)
//     - A tile: global->LDS via global_load_async_to_lds_b128 (ASYNCcnt)
//     - B tile: mask-free, lane-coalesced global_load_u16 gather (L2-resident)
//     - 8x v_wmma_f32_16x16x32_bf16 per wave per K-step, ping-pong LDS,
//       one barrier per step.
// Fallback (ws too small): round-2 self-contained kernel.
// ---------------------------------------------------------------------------

typedef __attribute__((ext_vector_type(16))) __bf16          v16bf;
typedef __attribute__((ext_vector_type(8)))  __bf16          v8bf;
typedef __attribute__((ext_vector_type(8)))  float           v8f;
typedef __attribute__((ext_vector_type(4)))  float           v4f;
typedef __attribute__((ext_vector_type(4)))  unsigned int    v4u;
typedef __attribute__((ext_vector_type(4)))  unsigned short  us4;

namespace {
constexpr int kCin  = 128;
constexpr int kH    = 56;
constexpr int kW    = 56;
constexpr int kCout = 256;
constexpr int kHW   = kH * kW;            // 3136
constexpr int kKdim = kCin * 9;           // 1152
constexpr int kNcol = 32 * kHW;           // 100352

constexpr int kHP   = 58;                 // padded H/W
constexpr int kHWP  = kHP * kHP;          // 3364
constexpr int kCHWP = kCin * kHWP;        // 430592 (per image, padded, elems)

constexpr int BM = 128;
constexpr int BN = 128;
constexpr int BK = 32;
constexpr int BK_PAD = 40;
constexpr int KTILES = kKdim / BK;        // 36

constexpr int kWpElems = kCout * 9 * kCin;        // 294912
constexpr int kXpElems = 32 * kCHWP;              // 13778944
} // namespace

__device__ __forceinline__ unsigned short f32_to_bf16_rne(float f) {
    unsigned u = __builtin_bit_cast(unsigned, f);
    u += 0x7FFFu + ((u >> 16) & 1u);
    return (unsigned short)(u >> 16);
}

// ===========================================================================
// prep kernels (one-time, bandwidth-trivial)
// ===========================================================================
__global__ __launch_bounds__(256)
void prep_weight(const float* __restrict__ wgt, unsigned short* __restrict__ wp) {
    const int idx = blockIdx.x * 256 + threadIdx.x;        // [co][tap][ci]
    const int co  = idx / kKdim;
    const int rem = idx - co * kKdim;
    const int tap = rem >> 7;
    const int ci  = rem & 127;
    wp[idx] = f32_to_bf16_rne(wgt[co * kKdim + ci * 9 + tap]);
}

__global__ __launch_bounds__(256)
void prep_x(const float* __restrict__ x, unsigned short* __restrict__ xp) {
    const int idx = blockIdx.x * 256 + threadIdx.x;        // [n*ci][58][58]
    const int nc  = idx / kHWP;
    const int hw  = idx - nc * kHWP;
    const int h   = hw / kHP;
    const int w   = hw - h * kHP;
    const int hi  = h - 1, wi = w - 1;
    float v = 0.0f;
    if ((unsigned)hi < (unsigned)kH && (unsigned)wi < (unsigned)kW)
        v = x[nc * kHW + hi * kW + wi];
    xp[idx] = f32_to_bf16_rne(v);
}

// ===========================================================================
// main kernel (fast path)
// ===========================================================================
__global__ __launch_bounds__(256)
void conv_main(const unsigned short* __restrict__ wp,
               const unsigned short* __restrict__ xp,
               const float* __restrict__ bias,
               float* __restrict__ out)
{
    __shared__ unsigned short As[2][BM][BK_PAD];   // weight tile [m][k] bf16
    __shared__ unsigned short Bs[2][BN][BK_PAD];   // im2col tile [n][k] bf16

    const int tid  = threadIdx.x;
    const int lane = tid & 31;
    const int wave = tid >> 5;

    const int block_m = blockIdx.y * BM;
    const int block_n = blockIdx.x * BN;

    // ---- A staging map: thread -> row m=t>>1, 16 consecutive k (32 B) -----
    const int a_m  = tid >> 1;
    const int a_kb = (tid & 1) * 16;
    const unsigned short* a_gbase = wp + (size_t)(block_m + a_m) * kKdim + a_kb;

    // ---- B staging map: lane -> pixel (coalesced), wave -> 4 k-values -----
    // thread covers pixels {block_n + lane + 32j, j=0..3} x k {wave*4+i, i=0..3}
    const int w4    = wave * 4;
    const unsigned w4off = (unsigned)(w4 * kHWP);
    unsigned pb2[4];                     // hoisted padded-base per pixel strip
#pragma unroll
    for (int j = 0; j < 4; ++j) {
        const int p    = block_n + lane + 32 * j;
        const int nimg = p / kHW;
        const int hw   = p - nimg * kHW;
        const int oh   = hw / kW;
        // padded offset oh*58+ow == hw + 2*oh
        pb2[j] = (unsigned)(nimg * kCHWP + hw + 2 * oh) + w4off;
    }

    // ---- wave tile position ----------------------------------------------
    const int wm     = (wave & 1) * 64;
    const int wn     = (wave >> 1) * 32;
    const int l16    = lane & 15;
    const int lhalf  = lane >> 4;
    const int a_koff = lhalf ? 8 : 0;
    const int b_koff = lhalf ? 8 : 0;

    // ---- async A tile: global -> LDS, no VGPR round-trip ------------------
    auto stage_a_async = [&](int buf, int kt) {
        const unsigned short* src = a_gbase + kt * BK;
        const unsigned ldsa =
            (unsigned)(unsigned long long)&As[buf][a_m][a_kb];
        asm volatile(
            "global_load_async_to_lds_b128 %0, %1, off\n\t"
            "global_load_async_to_lds_b128 %0, %1, off offset:16"
            :
            : "v"(ldsa), "v"(src)
            : "memory");
    };

    // ---- B gather: mask-free, lane-coalesced u16 loads --------------------
    auto stage_b_load = [&](int kt, unsigned short tv[16]) {
        const int tap = kt >> 2;                       // k = tap*128 + ci
        const int r   = tap / 3;
        const int s   = tap - r * 3;
        const int scl = ((kt & 3) << 5) * kHWP + r * kHP + s;   // SGPR
#pragma unroll
        for (int j = 0; j < 4; ++j) {
            const unsigned base = pb2[j] + (unsigned)scl;       // 1 v_add
#pragma unroll
            for (int i = 0; i < 4; ++i)
                tv[j * 4 + i] = xp[base + i * kHWP];   // imm-offset u16 loads
        }
    };
    auto stage_b_store = [&](int buf, const unsigned short tv[16]) {
#pragma unroll
        for (int j = 0; j < 4; ++j)
            *(us4*)&Bs[buf][lane + 32 * j][w4] =
                (us4){tv[4 * j], tv[4 * j + 1], tv[4 * j + 2], tv[4 * j + 3]};
    };

    // ---- accumulators -----------------------------------------------------
    v8f acc[4][2];
#pragma unroll
    for (int i = 0; i < 4; ++i)
#pragma unroll
        for (int j = 0; j < 2; ++j)
            acc[i][j] = v8f{};

    // ---- pipeline prologue ------------------------------------------------
    unsigned short tv[16];
    stage_a_async(0, 0);
    stage_b_load(0, tv);
    stage_b_store(0, tv);
    asm volatile("s_wait_asynccnt 0" ::: "memory");
    __syncthreads();

    // ---- main loop: one barrier per K-step, ping-pong buffers -------------
    for (int kt = 0; kt < KTILES; ++kt) {
        const int cur = kt & 1;

        if (kt + 1 < KTILES) {
            stage_a_async(cur ^ 1, kt + 1);      // ASYNCcnt copy under WMMA
            stage_b_load(kt + 1, tv);            // VMEM in flight under WMMA
        }

        v16bf afrag[4];
#pragma unroll
        for (int i = 0; i < 4; ++i) {
            const unsigned short* ap = &As[cur][wm + i * 16 + l16][0];
            v8bf clo = *(const v8bf*)(ap + a_koff);
            v8bf chi = *(const v8bf*)(ap + a_koff + 16);
            afrag[i] = __builtin_shufflevector(clo, chi,
                0,1,2,3,4,5,6,7,8,9,10,11,12,13,14,15);
        }
        v16bf bfrag[2];
#pragma unroll
        for (int j = 0; j < 2; ++j) {
            const unsigned short* bp = &Bs[cur][wn + j * 16 + l16][0];
            v8bf clo = *(const v8bf*)(bp + b_koff);
            v8bf chi = *(const v8bf*)(bp + b_koff + 16);
            bfrag[j] = __builtin_shufflevector(clo, chi,
                0,1,2,3,4,5,6,7,8,9,10,11,12,13,14,15);
        }

#pragma unroll
        for (int i = 0; i < 4; ++i)
#pragma unroll
            for (int j = 0; j < 2; ++j)
                acc[i][j] = __builtin_amdgcn_wmma_f32_16x16x32_bf16(
                    false, afrag[i], false, bfrag[j],
                    (short)0, acc[i][j], false, false);

        if (kt + 1 < KTILES)
            stage_b_store(cur ^ 1, tv);

        asm volatile("s_wait_asynccnt 0" ::: "memory");
        __syncthreads();
    }

    // ---- epilogue: bias add + scatter to NCHW -----------------------------
#pragma unroll
    for (int i = 0; i < 4; ++i) {
        const int co_base = block_m + wm + i * 16 + lhalf * 8;
#pragma unroll
        for (int j = 0; j < 2; ++j) {
            const int p    = block_n + wn + j * 16 + l16;
            const int nimg = p / kHW;
            const int hw   = p - nimg * kHW;
            float* obase = out + (size_t)nimg * (kCout * kHW) + hw;
#pragma unroll
            for (int v = 0; v < 8; ++v) {
                const int co = co_base + v;
                obase[(size_t)co * kHW] = acc[i][j][v] + bias[co];
            }
        }
    }
}

// ===========================================================================
// fallback (round-2 self-contained kernel, used only if ws_size too small)
// ===========================================================================
__global__ __launch_bounds__(256)
void conv_fallback(const float* __restrict__ x, const float* __restrict__ wgt,
                   const float* __restrict__ bias, float* __restrict__ out)
{
    __shared__ unsigned short As[2][BM][BK_PAD];
    __shared__ unsigned short Bs[2][BN][BK_PAD];

    const int tid  = threadIdx.x;
    const int lane = tid & 31;
    const int wave = tid >> 5;
    const int block_m = blockIdx.y * BM;
    const int block_n = blockIdx.x * BN;

    const int a_m  = tid >> 1;
    const int a_kb = (tid & 1) * 16;
    const float* a_gbase = wgt + (size_t)(block_m + a_m) * kKdim + a_kb;

    const int b_k  = tid & 31;
    const int b_nb = (tid >> 5) * 16;
    const int p0   = block_n + b_nb;

    unsigned boff[16];
    int      bpm[16];
#pragma unroll
    for (int i = 0; i < 16; ++i) {
        const int p = p0 + i;
        const int nimg = p / kHW;
        const int hw   = p - nimg * kHW;
        const int oh   = hw / kW;
        const int ow   = hw - oh * kW;
        boff[i] = (unsigned)(nimg * (kCin * kHW) + hw);
        int pm = 0;
#pragma unroll
        for (int r = 0; r < 3; ++r) {
            const bool hv = (unsigned)(oh + r - 1) < (unsigned)kH;
#pragma unroll
            for (int s = 0; s < 3; ++s) {
                const bool wv = (unsigned)(ow + s - 1) < (unsigned)kW;
                if (hv && wv) pm |= 1 << (r * 3 + s);
            }
        }
        bpm[i] = pm;
    }

    const int wm = (wave & 1) * 64, wn = (wave >> 1) * 32;
    const int l16 = lane & 15, lhalf = lane >> 4;
    const int a_koff = lhalf ? 8 : 0, b_koff = lhalf ? 8 : 0;

    auto stage_global = [&](int kt, v4f A[4], float bv[16]) {
        const float* ap = a_gbase + kt * BK;
        A[0] = *(const v4f*)(ap + 0);  A[1] = *(const v4f*)(ap + 4);
        A[2] = *(const v4f*)(ap + 8);  A[3] = *(const v4f*)(ap + 12);
        const int kg = kt * BK + b_k;
        const int ci = kg / 9, rs = kg - ci * 9;
        const int r = rs / 3, s = rs - r * 3;
        const int koff = ci * kHW + (r - 1) * kW + (s - 1);
        const unsigned bit = 1u << rs;
#pragma unroll
        for (int i = 0; i < 16; ++i) {
            const bool ok = (bpm[i] & bit) != 0;
            const int idx = ok ? (int)boff[i] + koff : 0;
            const float t = x[idx];
            bv[i] = ok ? t : 0.0f;
        }
    };
    auto stage_to_lds = [&](int buf, const v4f A[4], const float bv[16]) {
        const float* af = (const float*)A;
        unsigned pk[8];
#pragma unroll
        for (int i = 0; i < 8; ++i) {
            const unsigned lo = __builtin_bit_cast(unsigned, af[2*i])   + 0x8000u;
            const unsigned hi = __builtin_bit_cast(unsigned, af[2*i+1]) + 0x8000u;
            pk[i] = __builtin_amdgcn_perm(hi, lo, 0x07060302u);
        }
        *(v4u*)&As[buf][a_m][a_kb]     = (v4u){pk[0], pk[1], pk[2], pk[3]};
        *(v4u*)&As[buf][a_m][a_kb + 8] = (v4u){pk[4], pk[5], pk[6], pk[7]};
#pragma unroll
        for (int i = 0; i < 16; ++i) {
            const unsigned u = __builtin_bit_cast(unsigned, bv[i]) + 0x8000u;
            Bs[buf][b_nb + i][b_k] = (unsigned short)(u >> 16);
        }
    };

    v8f acc[4][2];
#pragma unroll
    for (int i = 0; i < 4; ++i)
#pragma unroll
        for (int j = 0; j < 2; ++j) acc[i][j] = v8f{};

    v4f Ar[4]; float bvr[16];
    stage_global(0, Ar, bvr);
    stage_to_lds(0, Ar, bvr);
    __syncthreads();

    for (int kt = 0; kt < KTILES; ++kt) {
        const int cur = kt & 1;
        if (kt + 1 < KTILES) stage_global(kt + 1, Ar, bvr);

        v16bf afrag[4];
#pragma unroll
        for (int i = 0; i < 4; ++i) {
            const unsigned short* ap = &As[cur][wm + i * 16 + l16][0];
            v8bf clo = *(const v8bf*)(ap + a_koff);
            v8bf chi = *(const v8bf*)(ap + a_koff + 16);
            afrag[i] = __builtin_shufflevector(clo, chi,
                0,1,2,3,4,5,6,7,8,9,10,11,12,13,14,15);
        }
        v16bf bfrag[2];
#pragma unroll
        for (int j = 0; j < 2; ++j) {
            const unsigned short* bp = &Bs[cur][wn + j * 16 + l16][0];
            v8bf clo = *(const v8bf*)(bp + b_koff);
            v8bf chi = *(const v8bf*)(bp + b_koff + 16);
            bfrag[j] = __builtin_shufflevector(clo, chi,
                0,1,2,3,4,5,6,7,8,9,10,11,12,13,14,15);
        }
#pragma unroll
        for (int i = 0; i < 4; ++i)
#pragma unroll
            for (int j = 0; j < 2; ++j)
                acc[i][j] = __builtin_amdgcn_wmma_f32_16x16x32_bf16(
                    false, afrag[i], false, bfrag[j],
                    (short)0, acc[i][j], false, false);

        if (kt + 1 < KTILES) stage_to_lds(cur ^ 1, Ar, bvr);
        __syncthreads();
    }

#pragma unroll
    for (int i = 0; i < 4; ++i) {
        const int co_base = block_m + wm + i * 16 + lhalf * 8;
#pragma unroll
        for (int j = 0; j < 2; ++j) {
            const int p = block_n + wn + j * 16 + l16;
            const int nimg = p / kHW;
            const int hw   = p - nimg * kHW;
            float* obase = out + (size_t)nimg * (kCout * kHW) + hw;
#pragma unroll
            for (int v = 0; v < 8; ++v) {
                const int co = co_base + v;
                obase[(size_t)co * kHW] = acc[i][j][v] + bias[co];
            }
        }
    }
}

// ===========================================================================
extern "C" void kernel_launch(void* const* d_in, const int* in_sizes, int n_in,
                              void* d_out, int out_size, void* d_ws, size_t ws_size,
                              hipStream_t stream) {
    const float* x    = (const float*)d_in[0];
    const float* wgt  = (const float*)d_in[1];
    const float* bias = (const float*)d_in[2];
    float* out        = (float*)d_out;

    dim3 grid(kNcol / BN, kCout / BM);   // (784, 2)
    dim3 block(256);

    const size_t need = (size_t)(kWpElems + kXpElems) * sizeof(unsigned short);
    if (ws_size >= need) {
        unsigned short* wp = (unsigned short*)d_ws;
        unsigned short* xp = wp + kWpElems;
        prep_weight<<<kWpElems / 256, 256, 0, stream>>>(wgt, wp);
        prep_x<<<kXpElems / 256, 256, 0, stream>>>(x, xp);
        conv_main<<<grid, block, 0, stream>>>(wp, xp, bias, out);
    } else {
        conv_fallback<<<grid, block, 0, stream>>>(x, wgt, bias, out);
    }
}
